// BoundaryPredictor1_69252052681053
// MI455X (gfx1250) — matrix operand order repaired and checked by reading, not compiled, not run on hardware
//
#include <hip/hip_runtime.h>

// ---------------------------------------------------------------------------
// BoundaryPredictor forward for MI455X (gfx1250, wave32, WMMA).
// B=8, L=2048, D=1024, H=2048.
//  K1 prep_bfrag : W1 f32 -> split-bf16 (hi,lo) packed in WMMA B-fragment layout
//  K2 logits     : fused relu(h@W1+b1)@W2+b2 via v_wmma_f32_16x16x32_bf16 x3
//                  (split-bf16 fp32 emulation), A tile (64 tok x 1024 K, hi+lo)
//                  staged in 256KB LDS (CDNA5: 320KB/WGP). Unroll capped to
//                  avoid the round-1 VGPR blowup + scratch spills.
//  K3 scan       : per-batch sequential boundary scan, segment extents,
//                  binomial NLL (lgammaf), shortened mask. Deterministic.
//  K4 pool       : segment mean-pool + sinusoidal pos-emb, HBM-roofline pass.
// ---------------------------------------------------------------------------

#define BB 8
#define LL 2048
#define DD 1024
#define HH 2048
#define TOK (BB * LL)          // 16384
#define HC (HH / 16)           // 128 H-chunks of 16 cols
#define KC (DD / 32)           // 32 K-chunks of 32
#define TILE_TOK 64            // tokens per block in logits kernel

typedef __bf16 bf16t;
typedef __attribute__((ext_vector_type(16))) __bf16 v16bf;
typedef __attribute__((ext_vector_type(8)))  float  v8f;

union Frag16 { v16bf v; uint4 q[2]; };
union Pack16 { unsigned short u[16]; uint4 q[2]; };

__device__ __forceinline__ unsigned short f2bf(float f) {
  unsigned u = __float_as_uint(f);
  u += 0x7fffu + ((u >> 16) & 1u);        // round-to-nearest-even
  return (unsigned short)(u >> 16);
}
__device__ __forceinline__ float bf2f(unsigned short h) {
  return __uint_as_float(((unsigned)h) << 16);
}

// --------------------------------------------------------------------------
// K1: pack W1 into WMMA 16-bit B-matrix (32x16) fragment layout, split hi/lo.
// Fragment f = h*KC + c ; per lane 16 bf16 values, stored contiguously so the
// GEMM kernel does two coalesced b128 loads per operand.
// Layout (dense 32x16 bf16 B): lanes 0-15 -> K=c*32+0..15 (col n = lane),
//                              lanes 16-31 -> K=c*32+16..31 (col n = lane-16).
// --------------------------------------------------------------------------
__global__ void prep_bfrag(const float* __restrict__ W1,
                           unsigned short* __restrict__ bhi,
                           unsigned short* __restrict__ blo) {
  int f = blockIdx.x;                     // 0..HC*KC-1
  int l = threadIdx.x;                    // 0..31
  int h = f >> 5, c = f & 31;
  int n = (h << 4) + (l & 15);
  int kbase = (c << 5) + ((l < 16) ? 0 : 16);
  Pack16 ph, pl;
#pragma unroll
  for (int i = 0; i < 16; ++i) {
    float x = W1[(size_t)(kbase + i) * HH + n];
    unsigned short xh = f2bf(x);
    ph.u[i] = xh;
    pl.u[i] = f2bf(x - bf2f(xh));
  }
  size_t off = ((size_t)f * 32 + l) * 16;
  ((uint4*)(bhi + off))[0] = ph.q[0];
  ((uint4*)(bhi + off))[1] = ph.q[1];
  ((uint4*)(blo + off))[0] = pl.q[0];
  ((uint4*)(blo + off))[1] = pl.q[1];
}

// --------------------------------------------------------------------------
// K2: logits = relu(hidden@W1 + b1)@W2 + b2 for 64 tokens per block.
// 8 waves: wave w handles M-subtile (w&3) and H-half (w>>2).
// Split-bf16 fp32 emulation: acc += Ahi*Bhi + Ahi*Blo + Alo*Bhi.
// Unroll factors pinned: j rolled, c unrolled x4 -> no VGPR-MSB/scratch spills,
// A fragments stream from LDS (ds_load_b128) each iteration as designed.
// --------------------------------------------------------------------------
__global__ __launch_bounds__(256) void logits_kernel(
    const float* __restrict__ hidden, const float* __restrict__ b1,
    const float* __restrict__ W2, const float* __restrict__ b2,
    const unsigned short* __restrict__ bhi, const unsigned short* __restrict__ blo,
    float* __restrict__ logits) {
  extern __shared__ unsigned short smem[];
  unsigned short* Ahi = smem;                      // 64*1024 ushort = 128KB
  unsigned short* Alo = smem + TILE_TOK * DD;      // 128KB
  float* partial = (float*)(smem + 2 * TILE_TOK * DD);  // [8][16] floats

  const int tid = threadIdx.x;
  const int tbase = blockIdx.x * TILE_TOK;

  // ---- Phase 1: stage A tile into LDS in 16-bit A-matrix (16x32) layout ----
  // lane = m + 16*(kin>>4), idx = kin&15
  for (int e = tid; e < TILE_TOK * DD; e += 256) {
    int tt = e >> 10;                 // local token 0..63
    int k  = e & (DD - 1);
    float x = hidden[(size_t)(tbase + tt) * DD + k];
    unsigned short xh = f2bf(x);
    unsigned short xl = f2bf(x - bf2f(xh));
    int mt = tt >> 4, m = tt & 15;
    int c = k >> 5, kin = k & 31;
    int lane = m + ((kin >> 4) << 4);
    int idx  = kin & 15;
    int off = (((mt * 32 + c) * 32 + lane) << 4) + idx;
    Ahi[off] = xh;
    Alo[off] = xl;
  }
  __syncthreads();

  const int w    = tid >> 5;          // wave 0..7
  const int lane = tid & 31;
  const int mt   = w & 3;             // M-subtile (16 tokens)
  const int half = w >> 2;            // H half
  const unsigned short* abh = Ahi + (size_t)mt * (32 * 32 * 16);
  const unsigned short* abl = Alo + (size_t)mt * (32 * 32 * 16);

  float lsum[8] = {0.f, 0.f, 0.f, 0.f, 0.f, 0.f, 0.f, 0.f};

#pragma unroll 1
  for (int j = 0; j < HC / 2; ++j) {  // 64 H-chunks per wave
    const int h = half * (HC / 2) + j;
    const int n = (h << 4) + (lane & 15);
    const float bias = b1[n];
    const float w2v  = W2[n];
    const unsigned short* bhbase = bhi + (size_t)h * 32 * 512;  // 32 frags * 32 lanes * 16
    const unsigned short* blbase = blo + (size_t)h * 32 * 512;
    v8f acc = {};
#pragma unroll 4
    for (int c = 0; c < KC; ++c) {
      Frag16 ah, al, fbh, fbl;
      const uint4* p0 = (const uint4*)(abh + ((c * 32 + lane) << 4));
      ah.q[0] = p0[0]; ah.q[1] = p0[1];
      const uint4* p1 = (const uint4*)(abl + ((c * 32 + lane) << 4));
      al.q[0] = p1[0]; al.q[1] = p1[1];
      const uint4* p2 = (const uint4*)(bhbase + ((c * 32 + lane) << 4));
      fbh.q[0] = p2[0]; fbh.q[1] = p2[1];
      const uint4* p3 = (const uint4*)(blbase + ((c * 32 + lane) << 4));
      fbl.q[0] = p3[0]; fbl.q[1] = p3[1];
      // speculative prefetch one unroll-body (4 c-steps) ahead; OOB is dropped
      __builtin_prefetch(bhbase + (((c + 4) * 32 + lane) << 4), 0, 3);
      __builtin_prefetch(blbase + (((c + 4) * 32 + lane) << 4), 0, 3);
      // fp32-accurate emulation: hi*hi + hi*lo + lo*hi (drop lo*lo)
      acc = __builtin_amdgcn_wmma_f32_16x16x32_bf16(false, ah.v, false, fbh.v,
                                                    (short)0, acc, false, false);
      acc = __builtin_amdgcn_wmma_f32_16x16x32_bf16(false, ah.v, false, fbl.v,
                                                    (short)0, acc, false, false);
      acc = __builtin_amdgcn_wmma_f32_16x16x32_bf16(false, al.v, false, fbh.v,
                                                    (short)0, acc, false, false);
    }
    // Fused ReLU + W2 weighting. C-layout: VGPR r, lanes<16 -> M=r, lanes>=16 -> M=r+8.
#pragma unroll
    for (int r = 0; r < 8; ++r) {
      float v = fmaxf(acc[r] + bias, 0.f);
      lsum[r] = fmaf(v, w2v, lsum[r]);
    }
  }

  // Reduce over the 16 N-lanes in each half-wave (stays within groups of 16).
#pragma unroll
  for (int r = 0; r < 8; ++r) {
    float v = lsum[r];
    v += __shfl_xor(v, 1, 32);
    v += __shfl_xor(v, 2, 32);
    v += __shfl_xor(v, 4, 32);
    v += __shfl_xor(v, 8, 32);
    lsum[r] = v;
  }
  if (lane == 0) {
#pragma unroll
    for (int r = 0; r < 8; ++r) partial[w * 16 + r] = lsum[r];       // rows 0..7
  }
  if (lane == 16) {
#pragma unroll
    for (int r = 0; r < 8; ++r) partial[w * 16 + 8 + r] = lsum[r];   // rows 8..15
  }
  __syncthreads();
  if (tid < TILE_TOK) {
    int pm = tid >> 4;                 // M-subtile
    int r  = tid & 15;                 // row in subtile
    float tot = partial[pm * 16 + r] + partial[(pm + 4) * 16 + r] + b2[0];
    logits[tbase + pm * 16 + r] = tot;
  }
}

// --------------------------------------------------------------------------
// K3: per-batch sequential scan (deterministic, tiny).
// --------------------------------------------------------------------------
__global__ void scan_kernel(const float* __restrict__ logits,
                            const float* __restrict__ mask,
                            const float* __restrict__ target,
                            int* __restrict__ seg_start, int* __restrict__ seg_len,
                            float* __restrict__ kc, float* __restrict__ nc,
                            float* __restrict__ out) {
  int b = threadIdx.x;
  if (b < BB) {
    const float* mrow = mask + b * LL;
    const float* lrow = logits + b * LL;
    // first pad index -> forced boundary at last real token
    int fp = -1;
    for (int l = 0; l < LL; ++l) { if (mrow[l] == 0.f) { fp = l; break; } }
    int lastreal = (fp >= 1) ? fp - 1 : -1;
    for (int s = 0; s < LL; ++s) seg_len[b * LL + s] = 0;
    int cum = 0, start = 0;
    float n = 0.f;
    for (int l = 0; l < LL; ++l) {
      float m = mrow[l];
      n += m;
      // sigmoid(logit) > 0.5  <=>  logit > 0
      int hb = (lrow[l] > 0.f && m != 0.f) ? 1 : 0;
      if (l == lastreal) hb = 1;
      if (hb) {
        seg_start[b * LL + cum] = start;
        seg_len[b * LL + cum] = l - start + 1;
        start = l + 1;
        ++cum;
      }
    }
    // full-length sequence with trailing unclosed segment
    if (fp < 0 && start < LL) {
      seg_start[b * LL + cum] = start;
      seg_len[b * LL + cum] = LL - start;
    }
    kc[b] = (float)cum;
    nc[b] = n;
    float* sm = out + (size_t)BB * LL * DD + 3 + (size_t)b * LL;
    for (int l = 0; l < LL; ++l) sm[l] = ((float)l < (float)cum) ? 1.f : 0.f;
  }
  __syncthreads();
  if (threadIdx.x == 0) {
    float loss = 0.f, nb = 0.f, tp = 0.f;
    for (int i = 0; i < BB; ++i) {
      float n = nc[i], k = kc[i];
      float p = target[i] / fmaxf(n, 1.f);
      p = fminf(fmaxf(p, 1e-6f), 1.f - 1e-6f);
      float lp = lgammaf(n + 1.f) - lgammaf(k + 1.f) - lgammaf(n - k + 1.f)
               + k * logf(p) + (n - k) * log1pf(-p);
      loss += -lp;
      nb += k;
      tp += n;
    }
    size_t base = (size_t)BB * LL * DD;
    out[base + 0] = 10.f * loss / (float)BB;
    out[base + 1] = nb;
    out[base + 2] = tp;
  }
}

// --------------------------------------------------------------------------
// K4: segment mean-pool + sinusoidal pos-emb (one HBM pass over hidden).
// --------------------------------------------------------------------------
__global__ __launch_bounds__(256) void pool_kernel(
    const float* __restrict__ hidden,
    const int* __restrict__ seg_start, const int* __restrict__ seg_len,
    float* __restrict__ pooled) {
  int blk = blockIdx.x;                  // 0..B*L-1
  int b = blk >> 11, s = blk & (LL - 1);
  int len = seg_len[b * LL + s];
  int start = (len > 0) ? seg_start[b * LL + s] : 0;
  float inv = 1.f / ((float)len + 1e-9f);
  const float* hp = hidden + (size_t)(b * LL + start) * DD;
  float* op = pooled + (size_t)(b * LL + s) * DD;
  const float LOG1E4 = 9.2103403719761836f;   // ln(10000)
  for (int d = threadIdx.x; d < DD; d += 256) {
    float sum = 0.f;
    for (int t = 0; t < len; ++t) sum += hp[(size_t)t * DD + d];
    float j2 = (float)(d & ~1);
    float ang = (float)s * expf(-(j2 * (1.f / (float)DD)) * LOG1E4);
    float pe = (d & 1) ? cosf(ang) : sinf(ang);
    op[d] = sum * inv + pe;
  }
}

// --------------------------------------------------------------------------
extern "C" void kernel_launch(void* const* d_in, const int* in_sizes, int n_in,
                              void* d_out, int out_size, void* d_ws, size_t ws_size,
                              hipStream_t stream) {
  const float* hidden = (const float*)d_in[0];
  const float* amask  = (const float*)d_in[1];
  const float* target = (const float*)d_in[2];
  const float* W1     = (const float*)d_in[3];
  const float* b1     = (const float*)d_in[4];
  const float* W2     = (const float*)d_in[5];
  const float* b2     = (const float*)d_in[6];
  float* out = (float*)d_out;

  char* ws = (char*)d_ws;
  unsigned short* bhi = (unsigned short*)ws;                         // 4 MB
  unsigned short* blo = (unsigned short*)(ws + (size_t)4194304);     // 4 MB
  float* logits       = (float*)(ws + (size_t)8388608);              // 64 KB
  int*   seg_start    = (int*)(ws + (size_t)8388608 + 65536);        // 64 KB
  int*   seg_len      = (int*)(ws + (size_t)8388608 + 131072);       // 64 KB
  float* kc           = (float*)(ws + (size_t)8388608 + 196608);     // 8
  float* nc           = kc + BB;                                     // 8

  // K1: W1 -> split-bf16 B fragments
  prep_bfrag<<<HC * KC, 32, 0, stream>>>(W1, bhi, blo);
  // K2: WMMA logits (256 blocks x 8 waves; 256KB+512B dynamic LDS per block)
  size_t lds_bytes = (size_t)2 * TILE_TOK * DD * sizeof(unsigned short)
                   + 8 * 16 * sizeof(float);
  logits_kernel<<<TOK / TILE_TOK, 256, lds_bytes, stream>>>(
      hidden, b1, W2, b2, bhi, blo, logits);
  // K3: boundary scan / loss / shortened mask
  scan_kernel<<<1, 32, 0, stream>>>(logits, amask, target, seg_start, seg_len,
                                    kc, nc, out);
  // K4: downsample + pos-emb -> pooled at out[0]
  pool_kernel<<<TOK, 256, 0, stream>>>(hidden, seg_start, seg_len, out);
}
